// ResACEUNet2_50560355008692
// MI455X (gfx1250) — matrix operand order repaired and checked by reading, not compiled, not run on hardware
//
#include <hip/hip_runtime.h>
#include <hip/hip_bf16.h>

// ---------------- problem constants ----------------
#define BB 4
#define CC 128
#define NN 32768           // H*W*D
#define MM (BB * NN)       // 131072 token rows
#define NH 4
#define DH 32
#define PROJ 64

typedef __attribute__((ext_vector_type(16))) _Float16 v16h;
typedef __attribute__((ext_vector_type(8)))  _Float16 v8h;
typedef __attribute__((ext_vector_type(8)))  float    v8f;

// ---------------- workspace layout (bytes) ----------------
#define OFF_QKVV   ((size_t)0)                        // f32 [M][512]           256 MB
#define OFF_FBUF   (OFF_QKVV + (size_t)MM*512*4)      // f16 [B][32][N][16]     128 MB
#define OFF_CONV   (OFF_FBUF + (size_t)BB*32*NN*16*2) // f32 [B][128][N]         64 MB
#define OFF_XH     (OFF_CONV + (size_t)MM*128*4)      // f16 [M][128]            32 MB
#define OFF_XSA    (OFF_XH   + (size_t)MM*128*2)      // f16 [M][128]            32 MB
#define OFF_XCA    (OFF_XSA  + (size_t)MM*128*2)      // f16 [M][128]            32 MB
#define OFF_ATOM   (OFF_XCA  + (size_t)MM*128*2)      // zero-init atomics region
#define OFF_SQSUM  (OFF_ATOM)                         // f32 [B][256]
#define OFF_S      (OFF_SQSUM + (size_t)1024*4)       // f32 [B][4][32][32]
#define OFF_KVPROJ (OFF_S     + (size_t)16384*4)      // f32 [2][B][128][64]
#define ATOM_FLOATS (1024 + 16384 + 65536)
#define OFF_INVN   (OFF_ATOM + (size_t)ATOM_FLOATS*4) // f32 [B][256]
#define OFF_ACA    (OFF_INVN + (size_t)1024*4)        // f32 [B][4][32][32]
#define OFF_WH     (OFF_ACA  + (size_t)16384*4)       // f16 Wqkvv [512][128]
#define OFF_WOH    (OFF_WH   + (size_t)512*128*2)     // f16 W_out [64][128]
#define OFF_WO2H   (OFF_WOH  + (size_t)64*128*2)      // f16 W_out2 [64][128]

// ---------------- WMMA fragment helpers (CDNA5 16x16x32 f16) ----------------
__device__ __forceinline__ v16h load_frag_a(const _Float16* __restrict__ A, int lda,
                                            int m0, int k0) {
  int l = threadIdx.x & 31;
  int row = m0 + (l & 15);
  int khalf = (l >> 4) * 16;
  const _Float16* p = A + (size_t)row * lda + k0 + khalf;
  v8h lo = *(const v8h*)(p);
  v8h hi = *(const v8h*)(p + 8);
  return __builtin_shufflevector(lo, hi, 0,1,2,3,4,5,6,7,8,9,10,11,12,13,14,15);
}

// B[k, col] = W[col, k] for row-major weight W [Nout][K]
__device__ __forceinline__ v16h load_frag_w(const _Float16* __restrict__ W, int ldw,
                                            int n0, int k0) {
  int l = threadIdx.x & 31;
  int col = n0 + (l & 15);
  int kb = k0 + ((l >> 4) << 4);
  return *(const v16h*)(W + (size_t)col * ldw + kb);
}

__device__ __forceinline__ void store_frag_d(float* __restrict__ D, int ldd,
                                             int m0, int n0, v8f c) {
  int l = threadIdx.x & 31;
  int row = m0 + ((l >> 4) << 3);
  int col = n0 + (l & 15);
#pragma unroll
  for (int r = 0; r < 8; ++r) D[(size_t)(row + r) * ldd + col] = c[r];
}

// ---------------- utility kernels ----------------
__global__ void k_zero_f32(float* __restrict__ p, int n) {
  int i = blockIdx.x * 256 + threadIdx.x;
  if (i < n) p[i] = 0.f;
}

__global__ void k_f32_to_f16(const float* __restrict__ s, _Float16* __restrict__ d, int n) {
  int i = blockIdx.x * 256 + threadIdx.x;
  if (i < n) d[i] = (_Float16)s[i];
}

// x [B][128][N] f32  ->  xh [B][N][128] f16  (LDS transpose tile 32c x 64n)
__global__ void k_pack_tokens(const float* __restrict__ x, _Float16* __restrict__ xh) {
  __shared__ float tile[32][65];
  int b = blockIdx.z, c0 = blockIdx.y * 32, n0 = blockIdx.x * 64;
  int t = threadIdx.x;
  for (int idx = t; idx < 32 * 64; idx += 256) {
    int cl = idx >> 6, nl = idx & 63;
    tile[cl][nl] = x[((size_t)(b * 128 + c0 + cl)) * NN + n0 + nl];
  }
  __syncthreads();
  for (int idx = t; idx < 64 * 32; idx += 256) {
    int nl = idx >> 5, cl = idx & 31;
    xh[((size_t)(b * NN + n0 + nl)) * 128 + c0 + cl] = (_Float16)tile[cl][nl];
  }
}

// ---------------- K1: qkvv = xf @ Wqkvv^T (WMMA) ----------------
// grid = M/16 blocks of 256 threads (8 waves); wave w covers out cols [w*64, w*64+64)
__global__ void k_gemm_qkvv(const _Float16* __restrict__ xh, const _Float16* __restrict__ wh,
                            float* __restrict__ qkvv) {
  int m0 = blockIdx.x * 16;
  int wave = threadIdx.x >> 5;
  int n0 = wave * 64;
  v8f acc0 = {}, acc1 = {}, acc2 = {}, acc3 = {};
#pragma unroll
  for (int k0 = 0; k0 < 128; k0 += 32) {
    v16h a = load_frag_a(xh, 128, m0, k0);
    v16h b0 = load_frag_w(wh, 128, n0 +  0, k0);
    v16h b1 = load_frag_w(wh, 128, n0 + 16, k0);
    v16h b2 = load_frag_w(wh, 128, n0 + 32, k0);
    v16h b3 = load_frag_w(wh, 128, n0 + 48, k0);
    acc0 = __builtin_amdgcn_wmma_f32_16x16x32_f16(false, a, false, b0, (short)0, acc0, false, false);
    acc1 = __builtin_amdgcn_wmma_f32_16x16x32_f16(false, a, false, b1, (short)0, acc1, false, false);
    acc2 = __builtin_amdgcn_wmma_f32_16x16x32_f16(false, a, false, b2, (short)0, acc2, false, false);
    acc3 = __builtin_amdgcn_wmma_f32_16x16x32_f16(false, a, false, b3, (short)0, acc3, false, false);
  }
  store_frag_d(qkvv, 512, m0, n0 +  0, acc0);
  store_frag_d(qkvv, 512, m0, n0 + 16, acc1);
  store_frag_d(qkvv, 512, m0, n0 + 32, acc2);
  store_frag_d(qkvv, 512, m0, n0 + 48, acc3);
}

// ---------------- K2a: 1x1 fc -> fbuf[b][d][n][o] (f16) ----------------
__global__ void k_fc(const float* __restrict__ qkvv, const float* __restrict__ Wfc,
                     const float* __restrict__ bfc, _Float16* __restrict__ fbuf) {
  __shared__ float sq[16][512];
  __shared__ float sw[16][16];
  __shared__ float sb[16];
  int b = blockIdx.y, n0 = blockIdx.x * 16;
  int t = threadIdx.x;
  if (t < 256) sw[t >> 4][t & 15] = Wfc[t];
  if (t < 16) sb[t] = bfc[t];
  for (int idx = t; idx < 16 * 512; idx += 256) {
    int tok = idx >> 9, c = idx & 511;
    sq[tok][c] = qkvv[((size_t)(b * NN + n0 + tok)) * 512 + c];
  }
  __syncthreads();
  int tok = t >> 4, o = t & 15;
  for (int d = 0; d < 32; ++d) {
    float acc = sb[o];
#pragma unroll
    for (int i = 0; i < 16; ++i) acc += sw[o][i] * sq[tok][i * 32 + d];
    fbuf[(((size_t)(b * 32 + d)) * NN + n0 + tok) * 16 + o] = (_Float16)acc;
  }
}

// ---------------- K2b: grouped 3x3x3 depthwise-ish conv ----------------
__global__ void k_dwconv(const _Float16* __restrict__ fbuf, const float* __restrict__ Wdep,
                         const float* __restrict__ bdep, float* __restrict__ convout) {
  __shared__ float wd[432];
  int b = blockIdx.z, oc = blockIdx.y, n0 = blockIdx.x * 256;
  int t = threadIdx.x;
  for (int idx = t; idx < 432; idx += 256) wd[idx] = Wdep[(size_t)oc * 432 + idx];
  __syncthreads();
  int n = n0 + t;
  int h = n >> 10, w = (n >> 5) & 31, dd = n & 31;
  int d = oc >> 2;
  const _Float16* fb = fbuf + ((size_t)(b * 32 + d)) * NN * 16;
  float acc = bdep[oc];
  for (int kz = -1; kz <= 1; ++kz) {
    int h2 = h + kz; if ((unsigned)h2 >= 32u) continue;
    for (int ky = -1; ky <= 1; ++ky) {
      int w2 = w + ky; if ((unsigned)w2 >= 32u) continue;
      for (int kx = -1; kx <= 1; ++kx) {
        int d2 = dd + kx; if ((unsigned)d2 >= 32u) continue;
        int kidx = (kz + 1) * 9 + (ky + 1) * 3 + (kx + 1);
        const _Float16* fp = fb + (size_t)((h2 << 10) | (w2 << 5) | d2) * 16;
#pragma unroll
        for (int i = 0; i < 16; ++i) acc += wd[i * 27 + kidx] * (float)fp[i];
      }
    }
  }
  convout[((size_t)(b * 128 + oc)) * NN + n] = acc;  // c-major [B][128][N]
}

// ---------------- K3: per-(b,channel) sum of squares for q,k rows ----------------
__global__ void k_sqsum(const float* __restrict__ qkvv, float* __restrict__ sqsum) {
  int b = blockIdx.y, n0 = blockIdx.x * 128;
  int t = threadIdx.x;  // channel 0..255 (q: 0..127, k: 128..255)
  float acc = 0.f;
  for (int j = 0; j < 128; ++j) {
    float v = qkvv[((size_t)(b * NN + n0 + j)) * 512 + t];
    acc += v * v;
  }
  atomicAdd(&sqsum[b * 256 + t], acc);
}

__global__ void k_invnorm(const float* __restrict__ sqsum, float* __restrict__ invn) {
  int i = blockIdx.x * 256 + threadIdx.x;
  if (i < 1024) {
    float s = sqrtf(sqsum[i]);
    invn[i] = 1.f / fmaxf(s, 1e-12f);
  }
}

// ---------------- K4: k_proj / v_SA_proj  (contract over N with W_E) ----------------
// thread t owns channel (t<128 ? k-ch 128+t : vSA-ch 256+t) and 64 register accumulators
__global__ void k_eproj(const float* __restrict__ qkvv, const float* __restrict__ WE,
                        float* __restrict__ kvproj) {
  __shared__ float swe[64][64];
  int b = blockIdx.y, n0 = blockIdx.x * 1024;
  int t = threadIdx.x;
  int ch = (t < 128) ? (128 + t) : (256 + t);
  float acc[64];
#pragma unroll
  for (int p = 0; p < 64; ++p) acc[p] = 0.f;
  for (int g = 0; g < 16; ++g) {
    int nb = n0 + g * 64;
    __syncthreads();
    for (int idx = t; idx < 64 * 64; idx += 256) {
      int p = idx >> 6, nl = idx & 63;
      swe[p][nl] = WE[(size_t)p * NN + nb + nl];
    }
    __syncthreads();
    for (int nl = 0; nl < 64; ++nl) {
      float v = qkvv[((size_t)(b * NN + nb + nl)) * 512 + ch];
#pragma unroll
      for (int p = 0; p < 64; ++p) acc[p] += v * swe[p][nl];
    }
  }
  float* dst = kvproj + ((t < 128) ? 0 : 32768) + ((size_t)b * 128 + (t & 127)) * 64;
#pragma unroll
  for (int p = 0; p < 64; ++p) atomicAdd(&dst[p], acc[p]);
}

__global__ void k_add_bE(float* __restrict__ kvproj, const float* __restrict__ bE) {
  int i = blockIdx.x * 256 + threadIdx.x;
  if (i < 65536) kvproj[i] += bE[i & 63];
}

// ---------------- K5: channel-attention gram S[b,h,d,e] = sum_n q*k ----------------
__global__ void k_gram(const float* __restrict__ qkvv, float* __restrict__ S) {
  __shared__ float sq[8][33], sk[8][33];
  int b = blockIdx.z, h = blockIdx.y, n0 = blockIdx.x * 1024;
  int t = threadIdx.x;
  int d = t >> 3, e0 = (t & 7) * 4;
  float a0 = 0, a1 = 0, a2 = 0, a3 = 0;
  for (int g = 0; g < 128; ++g) {
    int nb = n0 + g * 8;
    __syncthreads();
    {
      int tok = t >> 5, lane = t & 31;
      size_t base = ((size_t)(b * NN + nb + tok)) * 512 + h * 32 + lane;
      sq[tok][lane] = qkvv[base];
      sk[tok][lane] = qkvv[base + 128];
    }
    __syncthreads();
#pragma unroll
    for (int tok = 0; tok < 8; ++tok) {
      float qv = sq[tok][d];
      a0 += qv * sk[tok][e0 + 0];
      a1 += qv * sk[tok][e0 + 1];
      a2 += qv * sk[tok][e0 + 2];
      a3 += qv * sk[tok][e0 + 3];
    }
  }
  float* dst = &S[(((size_t)b * 4 + h) * 32 + d) * 32 + e0];
  atomicAdd(dst + 0, a0); atomicAdd(dst + 1, a1);
  atomicAdd(dst + 2, a2); atomicAdd(dst + 3, a3);
}

// ---------------- K5b: softmax over e with temperature + l2norm scaling ----------------
__global__ void k_softmax_ca(const float* __restrict__ S, const float* __restrict__ invn,
                             const float* __restrict__ temp, float* __restrict__ A) {
  __shared__ float s[32][33];
  int bh = blockIdx.x;          // b*4+h
  int b = bh >> 2, h = bh & 3;
  int t = threadIdx.x;
  float T = temp[h];
  for (int idx = t; idx < 1024; idx += 256) {
    int d = idx >> 5, e = idx & 31;
    float v = S[(size_t)bh * 1024 + idx];
    v *= invn[b * 256 + h * 32 + d] * invn[b * 256 + 128 + h * 32 + e] * T;
    s[d][e] = v;
  }
  __syncthreads();
  if (t < 32) {
    float m = -1e30f;
    for (int e = 0; e < 32; ++e) m = fmaxf(m, s[t][e]);
    float sum = 0.f;
    for (int e = 0; e < 32; ++e) sum += __expf(s[t][e] - m);
    float inv = 1.f / sum;
    for (int e = 0; e < 32; ++e)
      A[(size_t)bh * 1024 + t * 32 + e] = __expf(s[t][e] - m) * inv;
  }
}

// ---------------- K7: spatial (Linformer) attention, fused softmax ----------------
__global__ void k_spatial(const float* __restrict__ qkvv, const float* __restrict__ kvproj,
                          const float* __restrict__ invn, const float* __restrict__ temp2,
                          _Float16* __restrict__ xsa) {
  __shared__ float kp[32][65], vp[32][65];
  __shared__ float sq[4][33];
  __shared__ float sc[4][64];
  __shared__ float red[4][2];
  __shared__ float siq[32];
  int b = blockIdx.z, h = blockIdx.y, n0 = blockIdx.x * 64;
  int t = threadIdx.x;
  for (int idx = t; idx < 2048; idx += 256) {
    int d = idx >> 6, p = idx & 63;
    size_t base = ((size_t)b * 128 + h * 32 + d) * 64 + p;
    kp[d][p] = kvproj[base];
    vp[d][p] = kvproj[32768 + base];
  }
  if (t < 32) siq[t] = invn[b * 256 + h * 32 + t];
  float T2 = temp2[h];
  __syncthreads();
  for (int g = 0; g < 16; ++g) {
    int nb = n0 + g * 4;
    if (t < 128) {
      int tok = t >> 5, lane = t & 31;
      sq[tok][lane] = qkvv[((size_t)(b * NN + nb + tok)) * 512 + h * 32 + lane] * siq[lane];
    }
    __syncthreads();
    int tok = t >> 6, p = t & 63;
    float s = 0.f;
#pragma unroll
    for (int d = 0; d < 32; ++d) s += sq[tok][d] * kp[d][p];
    sc[tok][p] = s * T2;
    __syncthreads();
    if (t < 4) {
      float m = -1e30f;
      for (int pp = 0; pp < 64; ++pp) m = fmaxf(m, sc[t][pp]);
      float sum = 0.f;
      for (int pp = 0; pp < 64; ++pp) sum += __expf(sc[t][pp] - m);
      red[t][0] = m; red[t][1] = 1.f / sum;
    }
    __syncthreads();
    float soft = __expf(sc[tok][p] - red[tok][0]) * red[tok][1];
    __syncthreads();
    sc[tok][p] = soft;
    __syncthreads();
    if (p < 32) {
      int d = p;
      float o = 0.f;
#pragma unroll
      for (int pp = 0; pp < 64; ++pp) o += sc[tok][pp] * vp[d][pp];
      int n = nb + tok;
      // torch-faithful scrambled reshape: [b,h,n,d] -> (n' = (4d+h)*256 + n>>7, c' = n&127)
      int np = (4 * d + h) * 256 + (n >> 7);
      int cp = n & 127;
      xsa[((size_t)b * NN + np) * 128 + cp] = (_Float16)o;
    }
    __syncthreads();
  }
}

// ---------------- K8: x_CA = attn_CA @ v_CA   -> xca[b][n][c] f16 ----------------
__global__ void k_ca_apply(const float* __restrict__ qkvv, const float* __restrict__ A,
                           _Float16* __restrict__ xca) {
  __shared__ float sA[128][33];   // (h*32+d) x e
  __shared__ float vv[2][128];
  int b = blockIdx.y, n0 = blockIdx.x * 32;
  int t = threadIdx.x;
  for (int idx = t; idx < 4096; idx += 256) {
    int c = idx >> 5, e = idx & 31;
    sA[c][e] = A[((size_t)b * 4 + (c >> 5)) * 1024 + (c & 31) * 32 + e];
  }
  __syncthreads();
  int tok = t >> 7, c = t & 127;
  for (int g = 0; g < 16; ++g) {
    int n = n0 + g * 2 + tok;
    vv[tok][c] = qkvv[((size_t)(b * NN + n)) * 512 + 256 + c];
    __syncthreads();
    int h = c >> 5;
    float acc = 0.f;
#pragma unroll
    for (int e = 0; e < 32; ++e) acc += sA[c][e] * vv[tok][h * 32 + e];
    xca[((size_t)(b * NN + n)) * 128 + c] = (_Float16)acc;
    __syncthreads();
  }
}

// ---------------- K9: fused out projections (WMMA) + rate combine ----------------
// 8 waves: waves 0-3 -> x_SA @ W_out^T (cols 0..63), waves 4-7 -> x_CA @ W_out2^T (cols 64..127)
__global__ void k_gemm_out(const _Float16* __restrict__ xsa, const _Float16* __restrict__ xca,
                           const _Float16* __restrict__ wo, const _Float16* __restrict__ wo2,
                           const float* __restrict__ bo, const float* __restrict__ bo2,
                           const float* __restrict__ convout,
                           const float* __restrict__ rate, const float* __restrict__ rate2,
                           float* __restrict__ out) {
  int m0 = blockIdx.x * 16;
  int wave = threadIdx.x >> 5;
  const _Float16* Am = (wave < 4) ? xsa : xca;
  const _Float16* Wm = (wave < 4) ? wo : wo2;
  const float* bias  = (wave < 4) ? bo : bo2;
  int nt = (wave & 3) * 16;
  int ocol = ((wave < 4) ? 0 : 64) + nt;
  v8f acc = {};
#pragma unroll
  for (int k0 = 0; k0 < 128; k0 += 32) {
    v16h a = load_frag_a(Am, 128, m0, k0);
    v16h bfr = load_frag_w(Wm, 128, nt, k0);
    acc = __builtin_amdgcn_wmma_f32_16x16x32_f16(false, a, false, bfr, (short)0, acc, false, false);
  }
  float r1 = rate[0], r2 = rate2[0];
  int l = threadIdx.x & 31;
  int row = m0 + ((l >> 4) << 3);
  int col = ocol + (l & 15);
  float bv = bias[nt + (l & 15)];
#pragma unroll
  for (int r = 0; r < 8; ++r) {
    int m = row + r;
    int b = m >> 15;           // N = 32768
    int n = m & 32767;
    float cv = convout[((size_t)(b * 128 + col)) * NN + n];
    out[(size_t)m * 128 + col] = r1 * (acc[r] + bv) + r2 * cv;
  }
}

// ---------------- launcher ----------------
extern "C" void kernel_launch(void* const* d_in, const int* in_sizes, int n_in,
                              void* d_out, int out_size, void* d_ws, size_t ws_size,
                              hipStream_t stream) {
  (void)in_sizes; (void)n_in; (void)out_size; (void)ws_size;
  const float* x      = (const float*)d_in[0];
  const float* Wqkvv  = (const float*)d_in[1];
  const float* W_fc   = (const float*)d_in[2];
  const float* b_fc   = (const float*)d_in[3];
  const float* W_dep  = (const float*)d_in[4];
  const float* b_dep  = (const float*)d_in[5];
  const float* W_E    = (const float*)d_in[6];
  const float* b_E    = (const float*)d_in[7];
  const float* temp   = (const float*)d_in[8];
  const float* temp2  = (const float*)d_in[9];
  const float* rate   = (const float*)d_in[10];
  const float* rate2  = (const float*)d_in[11];
  const float* W_out  = (const float*)d_in[12];
  const float* b_out  = (const float*)d_in[13];
  const float* W_out2 = (const float*)d_in[14];
  const float* b_out2 = (const float*)d_in[15];
  float* out = (float*)d_out;

  char* ws = (char*)d_ws;
  float*    qkvv   = (float*)(ws + OFF_QKVV);
  _Float16* fbuf   = (_Float16*)(ws + OFF_FBUF);
  float*    convo  = (float*)(ws + OFF_CONV);
  _Float16* xh     = (_Float16*)(ws + OFF_XH);
  _Float16* xsa    = (_Float16*)(ws + OFF_XSA);
  _Float16* xca    = (_Float16*)(ws + OFF_XCA);
  float*    sqsum  = (float*)(ws + OFF_SQSUM);
  float*    Sgram  = (float*)(ws + OFF_S);
  float*    kvproj = (float*)(ws + OFF_KVPROJ);
  float*    invn   = (float*)(ws + OFF_INVN);
  float*    Aca    = (float*)(ws + OFF_ACA);
  _Float16* wh     = (_Float16*)(ws + OFF_WH);
  _Float16* woh    = (_Float16*)(ws + OFF_WOH);
  _Float16* wo2h   = (_Float16*)(ws + OFF_WO2H);

  // 0) zero atomic targets; convert weights; pack tokens
  k_zero_f32<<<(ATOM_FLOATS + 255) / 256, 256, 0, stream>>>((float*)(ws + OFF_ATOM), ATOM_FLOATS);
  k_f32_to_f16<<<(512 * 128 + 255) / 256, 256, 0, stream>>>(Wqkvv, wh, 512 * 128);
  k_f32_to_f16<<<(64 * 128 + 255) / 256, 256, 0, stream>>>(W_out, woh, 64 * 128);
  k_f32_to_f16<<<(64 * 128 + 255) / 256, 256, 0, stream>>>(W_out2, wo2h, 64 * 128);
  k_pack_tokens<<<dim3(NN / 64, 4, BB), 256, 0, stream>>>(x, xh);

  // 1) qkvv GEMM (WMMA)
  k_gemm_qkvv<<<MM / 16, 256, 0, stream>>>(xh, wh, qkvv);

  // 2) conv branch
  k_fc<<<dim3(NN / 16, BB), 256, 0, stream>>>(qkvv, W_fc, b_fc, fbuf);
  k_dwconv<<<dim3(NN / 256, 128, BB), 256, 0, stream>>>(fbuf, W_dep, b_dep, convo);

  // 3) l2 norms of q,k rows
  k_sqsum<<<dim3(NN / 128, BB), 256, 0, stream>>>(qkvv, sqsum);
  k_invnorm<<<4, 256, 0, stream>>>(sqsum, invn);

  // 4) Linformer E/F projections
  k_eproj<<<dim3(NN / 1024, BB), 256, 0, stream>>>(qkvv, W_E, kvproj);
  k_add_bE<<<65536 / 256, 256, 0, stream>>>(kvproj, b_E);

  // 5) channel attention
  k_gram<<<dim3(NN / 1024, NH, BB), 256, 0, stream>>>(qkvv, Sgram);
  k_softmax_ca<<<BB * NH, 256, 0, stream>>>(Sgram, invn, temp, Aca);
  k_ca_apply<<<dim3(NN / 32, BB), 256, 0, stream>>>(qkvv, Aca, xca);

  // 6) spatial attention (fused softmax)
  k_spatial<<<dim3(NN / 64, NH, BB), 256, 0, stream>>>(qkvv, kvproj, invn, temp2, xsa);

  // 7) fused output projections + rate combine (WMMA)
  k_gemm_out<<<MM / 16, 256, 0, stream>>>(xsa, xca, woh, wo2h, b_out, b_out2,
                                          convo, rate, rate2, out);
}
